// scaled_dot_product_attention_11493332484788
// MI455X (gfx1250) — compile-verified, matrix operand order checked by reading, not run on hardware
//
#include <hip/hip_runtime.h>
#include <hip/hip_bf16.h>
#include <math.h>

typedef __attribute__((ext_vector_type(16))) _Float16 v16h;
typedef __attribute__((ext_vector_type(8)))  _Float16 v8h;
typedef __attribute__((ext_vector_type(4)))  _Float16 v4h;
typedef __attribute__((ext_vector_type(8)))  float    v8f;
typedef __attribute__((ext_vector_type(4)))  float    f4;

#define MASK_FILL -1000000.0f
#define KS_STRIDE 72   // halves: 144 B row stride -> 16B-aligned frags, conflict-free reads
#define VT_STRIDE 40   // halves: 80 B row stride  -> 16B-aligned frags, conflict-free reads
#define KS_TILE   (32 * KS_STRIDE)
#define VT_TILE   (64 * VT_STRIDE)

// Per-lane WMMA fragment (16 halves) from an LDS row: 8 at off0, 8 at off0+16.
__device__ __forceinline__ v16h lds_frag(const _Float16* rowbase, int off0) {
  v8h x = *(const v8h*)(rowbase + off0);
  v8h y = *(const v8h*)(rowbase + off0 + 16);
  v16h r;
#pragma unroll
  for (int j = 0; j < 8; ++j) { r[j] = x[j]; r[j + 8] = y[j]; }
  return r;
}

// Q fragment from global f32, pre-scaled by (1/sqrt(D)) * log2(e): softmax runs in exp2 domain.
__device__ __forceinline__ v16h load_q_frag(const float* __restrict__ row,
                                            int chunk, int doff, float scale) {
  const float* base = row + chunk * 32 + doff;
  f4 a0 = *(const f4*)(base);
  f4 a1 = *(const f4*)(base + 4);
  f4 a2 = *(const f4*)(base + 16);
  f4 a3 = *(const f4*)(base + 20);
  v16h r;
#pragma unroll
  for (int i = 0; i < 4; ++i) {
    r[i]      = (_Float16)(a0[i] * scale);
    r[i + 4]  = (_Float16)(a1[i] * scale);
    r[i + 8]  = (_Float16)(a2[i] * scale);
    r[i + 12] = (_Float16)(a3[i] * scale);
  }
  return r;
}

// Issue global loads for one 32-key tile into registers (no conversion yet).
__device__ __forceinline__ void load_tile_regs(const float* __restrict__ Kb,
                                               const float* __restrict__ Vb,
                                               int k0, int tid, f4* kreg, f4* vreg) {
#pragma unroll
  for (int i = 0; i < 4; ++i) {
    int linear = tid + i * 128;        // 0..511 = 32 keys x 16 float4 cols
    int kr = linear >> 4;
    int c4 = linear & 15;
    kreg[i] = *(const f4*)(Kb + (size_t)(k0 + kr) * 64 + c4 * 4);
    vreg[i] = *(const f4*)(Vb + (size_t)(k0 + kr) * 64 + c4 * 4);
  }
}

// Convert registers to f16 and store: K row-major, V transposed.
__device__ __forceinline__ void store_tile(const f4* kreg, const f4* vreg,
                                           _Float16* Ks, _Float16* Vt, int tid) {
#pragma unroll
  for (int i = 0; i < 4; ++i) {
    int linear = tid + i * 128;
    int kr = linear >> 4;
    int c4 = linear & 15;
    v4h kh;
#pragma unroll
    for (int j = 0; j < 4; ++j) kh[j] = (_Float16)kreg[i][j];
    *(v4h*)(&Ks[kr * KS_STRIDE + c4 * 4]) = kh;
#pragma unroll
    for (int j = 0; j < 4; ++j)
      Vt[(c4 * 4 + j) * VT_STRIDE + kr] = (_Float16)vreg[i][j];
  }
}

__device__ __forceinline__ void process_tile(bool maskTile, int k0, int L,
    const _Float16* Ks, const _Float16* Vt,
    const v16h& qa0, const v16h& qa1,
    int lane16, int hi, int doff,
    v8f& ot0, v8f& ot1, v8f& ot2, v8f& ot3, float& m, float& l) {
  // Issue ALL LDS fragment loads up front: K frags feed the QK WMMAs immediately;
  // V frags (independent of the scores) hide their latency under QK + softmax.
  const _Float16* kr0 = Ks + (size_t)lane16 * KS_STRIDE;
  const _Float16* kr1 = Ks + (size_t)(16 + lane16) * KS_STRIDE;
  v16h ka00 = lds_frag(kr0, doff);
  v16h ka01 = lds_frag(kr0, 32 + doff);
  v16h ka10 = lds_frag(kr1, doff);
  v16h ka11 = lds_frag(kr1, 32 + doff);
  v16h va0  = lds_frag(Vt + (size_t)(0 * 16 + lane16) * VT_STRIDE, doff);
  v16h va1  = lds_frag(Vt + (size_t)(1 * 16 + lane16) * VT_STRIDE, doff);
  v16h va2  = lds_frag(Vt + (size_t)(2 * 16 + lane16) * VT_STRIDE, doff);
  v16h va3  = lds_frag(Vt + (size_t)(3 * 16 + lane16) * VT_STRIDE, doff);

  // S^T = K · Q^T (log2e-scaled), chained over D=64 in two K=32 WMMAs per 16-key subtile.
  v8f z = {};
  v8f s0 = __builtin_amdgcn_wmma_f32_16x16x32_f16(false, ka00, false, qa0, (short)0, z,  false, false);
  s0     = __builtin_amdgcn_wmma_f32_16x16x32_f16(false, ka01, false, qa1, (short)0, s0, false, false);
  v8f s1 = __builtin_amdgcn_wmma_f32_16x16x32_f16(false, ka10, false, qa0, (short)0, z,  false, false);
  s1     = __builtin_amdgcn_wmma_f32_16x16x32_f16(false, ka11, false, qa1, (short)0, s1, false, false);

  if (maskTile) {  // uniform scalar branch; taken once (final partial tile)
    const int kb0 = k0 + hi * 8;
    const int kb1 = k0 + 16 + hi * 8;
#pragma unroll
    for (int j = 0; j < 8; ++j) {
      if (kb0 + j >= L) s0[j] = MASK_FILL;
      if (kb1 + j >= L) s1[j] = MASK_FILL;
    }
  }

  // Online softmax in exp2 domain; one query per lane, merge the two lane-halves.
  float tmax = fmaxf(s0[0], s1[0]);
#pragma unroll
  for (int j = 1; j < 8; ++j) tmax = fmaxf(tmax, fmaxf(s0[j], s1[j]));
  tmax = fmaxf(tmax, __shfl_xor(tmax, 16, 32));
  const float nm   = fmaxf(m, tmax);
  const float corr = __builtin_amdgcn_exp2f(m - nm);
  m = nm;

#pragma unroll
  for (int j = 0; j < 8; ++j) {
    s0[j] = __builtin_amdgcn_exp2f(s0[j] - nm);
    s1[j] = __builtin_amdgcn_exp2f(s1[j] - nm);
  }
  float r = s0[0] + s1[0];
#pragma unroll
  for (int j = 1; j < 8; ++j) r += s0[j] + s1[j];
  r += __shfl_xor(r, 16, 32);
  l = l * corr + r;

  // S^T C-layout == B-operand layout of P^T (32 keys x 16 queries): pack in-lane.
  v16h pb;
#pragma unroll
  for (int j = 0; j < 8; ++j) { pb[j] = (_Float16)s0[j]; pb[j + 8] = (_Float16)s1[j]; }

#pragma unroll
  for (int j = 0; j < 8; ++j) { ot0[j] *= corr; ot1[j] *= corr; ot2[j] *= corr; ot3[j] *= corr; }

  // O^T += V^T · P^T; V frags already resident -> WMMAs issue back-to-back.
  ot0 = __builtin_amdgcn_wmma_f32_16x16x32_f16(false, va0, false, pb, (short)0, ot0, false, false);
  ot1 = __builtin_amdgcn_wmma_f32_16x16x32_f16(false, va1, false, pb, (short)0, ot1, false, false);
  ot2 = __builtin_amdgcn_wmma_f32_16x16x32_f16(false, va2, false, pb, (short)0, ot2, false, false);
  ot3 = __builtin_amdgcn_wmma_f32_16x16x32_f16(false, va3, false, pb, (short)0, ot3, false, false);
}

__global__ __launch_bounds__(128)
void fa_fwd_f16wmma_kernel(const float* __restrict__ Qg,
                           const float* __restrict__ Kg,
                           const float* __restrict__ Vg,
                           const int*   __restrict__ VL,
                           float*       __restrict__ Og) {
  constexpr int NQ = 2048, NK = 2048, D = 64;
  __shared__ __align__(16) _Float16 Ks[2 * KS_TILE];
  __shared__ __align__(16) _Float16 Vt[2 * VT_TILE];

  const int b      = blockIdx.y;
  const int tid    = threadIdx.x;
  const int lane   = tid & 31;
  const int wv     = tid >> 5;
  const int lane16 = lane & 15;
  const int hi     = (lane >> 4) & 1;
  const int doff   = hi * 8;

  const int L      = VL[b];
  const int nTiles = (L + 31) >> 5;
  const int rem    = L & 31;

  const float* Qb = Qg + (size_t)b * NQ * D;
  const float* Kb = Kg + (size_t)b * NK * D;
  const float* Vb = Vg + (size_t)b * NK * D;

  const int qr = blockIdx.x * 64 + wv * 16 + lane16;
  const float qscale = 0.125f * 1.44269504088896340736f;  // 1/sqrt(64) * log2(e)
  const v16h qa0 = load_q_frag(Qb + (size_t)qr * D, 0, doff, qscale);
  const v16h qa1 = load_q_frag(Qb + (size_t)qr * D, 1, doff, qscale);

  v8f ot0 = {}, ot1 = {}, ot2 = {}, ot3 = {};
  float m = -INFINITY, l = 0.0f;

  // Register-pipelined double-buffered loop: global loads for t+1 overlap WMMAs of t.
  f4 kreg[4], vreg[4];
  load_tile_regs(Kb, Vb, 0, tid, kreg, vreg);

  for (int t = 0; t < nTiles; ++t) {
    _Float16* Ksb = Ks + (t & 1) * KS_TILE;
    _Float16* Vtb = Vt + (t & 1) * VT_TILE;
    store_tile(kreg, vreg, Ksb, Vtb, tid);   // waits on t's loads, converts, stores
    __syncthreads();                         // staging of buffer (t&1) complete

    if (t + 1 < nTiles)
      load_tile_regs(Kb, Vb, (t + 1) * 32, tid, kreg, vreg);  // async w.r.t. compute
    const int kpf = (t + 2) * 32;
    if (kpf < NK) {                          // prime L2 two tiles ahead
      __builtin_prefetch(Kb + (size_t)kpf * D + tid * 16, 0, 0);
      __builtin_prefetch(Vb + (size_t)kpf * D + tid * 16, 0, 0);
    }

    process_tile(rem && (t == nTiles - 1), t * 32, L, Ksb, Vtb, qa0, qa1,
                 lane16, hi, doff, ot0, ot1, ot2, ot3, m, l);
  }

  const float rl = 1.0f / l;
  float* orow = Og + ((size_t)b * NQ + qr) * D;
#pragma unroll
  for (int g = 0; g < 4; ++g) {
    v8f o = (g == 0) ? ot0 : (g == 1) ? ot1 : (g == 2) ? ot2 : ot3;
    f4 lo  = { o[0] * rl, o[1] * rl, o[2] * rl, o[3] * rl };
    f4 hiv = { o[4] * rl, o[5] * rl, o[6] * rl, o[7] * rl };
    float* dst = orow + g * 16 + hi * 8;
    *(f4*)(dst)     = lo;
    *(f4*)(dst + 4) = hiv;
  }
}

extern "C" void kernel_launch(void* const* d_in, const int* in_sizes, int n_in,
                              void* d_out, int out_size, void* d_ws, size_t ws_size,
                              hipStream_t stream) {
  (void)in_sizes; (void)n_in; (void)out_size; (void)d_ws; (void)ws_size;
  const float* Q  = (const float*)d_in[0];
  const float* K  = (const float*)d_in[1];
  const float* V  = (const float*)d_in[2];
  const int*   VL = (const int*)d_in[3];
  float* O = (float*)d_out;

  dim3 grid(2048 / 64, 16);  // (query blocks of 64, batch)
  dim3 block(128);           // 4 waves, 16 queries each
  hipLaunchKernelGGL(fa_fwd_f16wmma_kernel, grid, block, 0, stream, Q, K, V, VL, O);
}
